// Encoder_42142219109010
// MI455X (gfx1250) — matrix operand order repaired and checked by reading, not compiled, not run on hardware
//
#include <hip/hip_runtime.h>
#include <cstdint>

// Problem constants (fixed by the reference: B=2, D=192).
#define DDIM 192
#define DD2  (DDIM * DDIM)          // 36864
#define DD3  (DDIM * DDIM * DDIM)   // 7,077,888
#define NVOX (2 * DD3)              // 14,155,776 voxels total (batch of 2)

typedef __attribute__((ext_vector_type(16))) _Float16 v16h;
typedef __attribute__((ext_vector_type(8)))  float    v8f;

union Frag16 { uint32_t u[8]; v16h h; };

// ---------------------------------------------------------------------------
// Kernel 0: reset the occupancy-list counter (ws is poisoned; we must be
// deterministic per call, so reset every launch).
// ---------------------------------------------------------------------------
__global__ __launch_bounds__(32) void k_reset(uint32_t* __restrict__ counter) {
    if (threadIdx.x == 0) counter[0] = 0u;
}

// ---------------------------------------------------------------------------
// Kernel 1: single streaming pass over `data`:
//   - d_out := 0 (prob and regr planes; masked-out voxels stay exactly 0)
//   - xA    := packed f16x2 {ch0 = data, ch1 = 0}  (layer-1 input, 1 channel)
//   - xB    := 0 (ping-pong buffer; sparse layers only write listed voxels)
//   - compact occupied voxel indices into `list` via wave ballot + atomic base
// Activation buffers have a guard dword at slot 0 (always zero); voxel v
// lives at slot v+1. Out-of-bounds gathers hit the guard -> exact zero with
// no post-load masking (keeps the gather loads dependency-free).
// ---------------------------------------------------------------------------
__global__ __launch_bounds__(256) void k_scan(
    const float* __restrict__ data,
    uint32_t* __restrict__ counter,
    uint32_t* __restrict__ list,
    uint32_t* __restrict__ xA,   // NVOX+1 dwords, guarded
    uint32_t* __restrict__ xB,   // NVOX+1 dwords, guarded
    float* __restrict__ out)
{
    const int i = blockIdx.x * blockDim.x + threadIdx.x;   // grid covers NVOX exactly
    const float d = data[i];
    const bool occ = (d != 0.0f);

    const unsigned short hb = __builtin_bit_cast(unsigned short, (_Float16)d);
    xA[i + 1] = (uint32_t)hb;    // {lo: f16(data), hi: 0} at guarded slot
    xB[i + 1] = 0u;
    out[i] = 0.0f;               // prob plane
    out[i + NVOX] = 0.0f;        // regr plane
    if (i == 0) { xA[0] = 0u; xB[0] = 0u; }   // guard slots

    const unsigned bal = (unsigned)__ballot(occ);          // wave32
    const int lane = threadIdx.x & 31;
    const int tot  = __popc(bal);
    unsigned base = 0u;
    if (lane == 0 && tot) base = atomicAdd(counter, (unsigned)tot);
    base = (unsigned)__shfl((int)base, 0, 32);
    if (occ) {
        const int pfx = __popc(bal & ((1u << lane) - 1u));
        list[base + pfx] = (uint32_t)i;
    }
}

// ---------------------------------------------------------------------------
// Generic sparse dilated 3x3x3 conv layer via V_WMMA_F32_16X16X32_F16.
//   GEMM:  M = 16 occupied voxels (from list), N = 2 output channels (cols 0,1
//          of the 16-wide tile), K = ci + 2*tap  (54 real, padded to 64 ->
//          two chained WMMA ops).
//   A (16x32 f16): lane l holds row M=l%16; lanes<16 carry K {0..7,16..23},
//          lanes>=16 carry K {8..15,24..31} (ISA 7.12.2). Each K-pair
//          (ci0,ci1 at one tap) is one interleaved-f16 dword gather.
//   B (32x16 f16): lane l holds col N=l%16; K = elem + 8*(l>=16) + 8*(elem>=8), same as A. Weights,
//          built once per wave.
//
//   Hot loop: branchless, 32-bit, and dependency-free after the load:
//     - per-lane tap geometry (delta+1, mask shift amount) hoisted out
//     - per-tile 27-bit SAME-padding validity mask m27; per tap a smear
//       (int)(m27 << shamt) >> 31 gates the *index only*; invalid gathers
//       collapse to the guard slot (zero), so loads feed WMMA directly and
//       can be issued back-to-back with counted s_wait_loadcnt.
//   Epilogue voxel indices come from ds_bpermute of the already-loaded list
//   entries (no global reloads / loadcnt stalls), stores use 32-bit offsets.
//
//   FINAL==0: epilogue = relu -> f16 scatter into xout (interleaved 2ch).
//   FINAL==1: relu(conv4) then fused 1x1x1 layer-5 (shfl partner channel),
//          sigmoid, scatter f32 into prob/regr.
// Mask multiply is a no-op: listed voxels have mask==1 by construction.
// ---------------------------------------------------------------------------
template <int FINAL>
__global__ __launch_bounds__(256) void k_conv(
    const uint32_t* __restrict__ xin,     // guarded base: slot v+1 = voxel v
    _Float16* __restrict__ xout,          // pre-biased by +1 dword (FINAL==0)
    float* __restrict__ prob,             // FINAL==1 destinations
    float* __restrict__ regr,
    const uint32_t* __restrict__ counter,
    const uint32_t* __restrict__ list,
    const float* __restrict__ W,          // (2, cin, 3, 3, 3) f32
    const float* __restrict__ W5,         // (2, 2) f32, FINAL only
    int cin, int dil)
{
    const int tid   = blockIdx.x * blockDim.x + threadIdx.x;
    const int lane  = threadIdx.x & 31;
    const int wave  = tid >> 5;
    const int nwav  = (gridDim.x * blockDim.x) >> 5;

    const int nCol  = lane & 15;     // N column (0..15); only 0,1 meaningful
    const int hi    = lane >> 4;     // lane-half select
    const int b2    = hi * 4;        // A: tap base per lane-half

    // ---- B fragments (weights), built once per wave -----------------------
    Frag16 bf[2];
#pragma unroll
    for (int c = 0; c < 2; ++c) {
#pragma unroll
        for (int e = 0; e < 16; ++e) {
            const int k   = e + 8 * hi + ((e >= 8) ? 8 : 0) + 32 * c;   // B-layout K index
            const int tap = k >> 1;
            const int ci  = k & 1;
            float w = 0.0f;
            if (nCol < 2 && tap < 27 && ci < cin)
                w = W[(nCol * cin + ci) * 27 + tap];
            bf[c].h[e] = (_Float16)w;
        }
    }
    float w5[4] = {0.f, 0.f, 0.f, 0.f};
    if constexpr (FINAL) { w5[0]=W5[0]; w5[1]=W5[1]; w5[2]=W5[2]; w5[3]=W5[3]; }

    // ---- per-lane tap geometry, hoisted out of the tile loop --------------
    // tap(c,j) = b2 + j + ((j>=4)?4:0) + 16*c   (lane-half dependent)
    int32_t dplus16[16];   // linear voxel offset of each tap, +1 (guard bias)
    int32_t sham16[16];    // 31 - tap (validity-bit smear shift)
#pragma unroll
    for (int c = 0; c < 2; ++c) {
#pragma unroll
        for (int j = 0; j < 8; ++j) {
            const int tap = b2 + j + ((j >= 4) ? 4 : 0) + 16 * c;  // 0..31
            const int tz  = tap / 9;
            const int rm  = tap - 9 * tz;
            const int ty  = rm / 3;
            const int tx  = rm - 3 * ty;
            dplus16[c * 8 + j] = (tz - 1) * dil * DD2 + (ty - 1) * dil * DDIM
                               + (tx - 1) * dil + 1;
            sham16[c * 8 + j]  = 31 - tap;
        }
    }

    const int count  = (int)counter[0];
    const int ntiles = (count + 15) >> 4;

    for (int t = wave; t < ntiles; t += nwav) {   // wave-uniform loop
        // --- my M-row voxel (rows duplicated across both lane halves) ------
        const int mi = t * 16 + nCol;
        const uint32_t vm = (mi < count) ? list[mi] : 0u;
        int tq = (int)vm;
        const int x = tq % DDIM; tq /= DDIM;
        const int y = tq % DDIM; tq /= DDIM;
        const int z = tq % DDIM;                  // batch never out of bounds

        // --- 27-bit SAME-padding validity mask for this voxel --------------
        const uint32_t mx = (x >= dil ? 1u : 0u) | 2u | (x + dil < DDIM ? 4u : 0u);
        const uint32_t my = (y >= dil ? 1u : 0u) | 2u | (y + dil < DDIM ? 4u : 0u);
        const uint32_t mz = (z >= dil ? 1u : 0u) | 2u | (z + dil < DDIM ? 4u : 0u);
        uint32_t m9  = ((my & 1u) ? mx : 0u) | ((my & 2u) ? (mx << 3) : 0u)
                     | ((my & 4u) ? (mx << 6) : 0u);
        uint32_t m27 = ((mz & 1u) ? m9 : 0u) | ((mz & 2u) ? (m9 << 9) : 0u)
                     | ((mz & 4u) ? (m9 << 18) : 0u);   // bits >= 27 are zero

        v8f acc = {};
#pragma unroll
        for (int c = 0; c < 2; ++c) {
            // Phase 1: all 8 gather indices (invalid taps -> 0 = guard slot).
            uint32_t idxs[8];
#pragma unroll
            for (int j = 0; j < 8; ++j) {
                const uint32_t sm =
                    (uint32_t)((int32_t)(m27 << sham16[c * 8 + j]) >> 31);
                idxs[j] = (vm + (uint32_t)dplus16[c * 8 + j]) & sm;
            }
            // Phase 2: independent loads feed the A fragment directly.
            Frag16 af;
#pragma unroll
            for (int j = 0; j < 8; ++j) af.u[j] = xin[idxs[j]];

            acc = __builtin_amdgcn_wmma_f32_16x16x32_f16(
                      false, af.h, false, bf[c].h, (short)0, acc, false, false);
        }

        // --- epilogue: lane l, VGPR r -> (M = r + 8*hi, N = nCol) ----------
        // Voxel index of row (hi*8+r) is lane (hi*8+r)'s vm: grab it with a
        // ds_bpermute instead of re-loading the list from global memory.
        if constexpr (FINAL) {
#pragma unroll
            for (int r = 0; r < 8; ++r) {
                const float v  = fmaxf(acc[r], 0.0f);        // relu(conv4)
                const float vo = __shfl(v, lane ^ 1, 32);    // partner channel
                const uint32_t vox = (uint32_t)__shfl((int)vm, hi * 8 + r, 32);
                const int msto = t * 16 + hi * 8 + r;
                if (nCol == 0 && msto < count) {
                    const float y0 = w5[0] * v + w5[1] * vo; // 1x1x1 layer 5
                    const float y1 = w5[2] * v + w5[3] * vo;
                    prob[vox] = 1.0f / (1.0f + __expf(-y0)); // sigmoid
                    regr[vox] = 1.0f / (1.0f + __expf(-y1));
                }
            }
        } else {
#pragma unroll
            for (int r = 0; r < 8; ++r) {
                const uint32_t vox = (uint32_t)__shfl((int)vm, hi * 8 + r, 32);
                const _Float16 hv = (_Float16)fmaxf(acc[r], 0.0f);
                const int msto = t * 16 + hi * 8 + r;
                if (nCol < 2 && msto < count) {
                    xout[vox * 2u + (uint32_t)nCol] = hv;    // 32-bit offset
                }
            }
        }
    }
}

// ---------------------------------------------------------------------------
// Host launcher. Workspace layout (requires ~170 MB of d_ws):
//   [0,256)                       : occupancy counter
//   [256, 256+4*NVOX)             : occupied-voxel index list (worst case)
//   [.., +4*(NVOX+1))             : xA  (guard dword + packed f16x2 per voxel)
//   [.., +4*(NVOX+1))             : xB  (guard dword + packed f16x2 per voxel)
// Ping-pong: L1 xA->xB, L2 xB->xA, L3 xA->xB, L4+L5 xB->d_out.
// ---------------------------------------------------------------------------
extern "C" void kernel_launch(void* const* d_in, const int* in_sizes, int n_in,
                              void* d_out, int out_size, void* d_ws, size_t ws_size,
                              hipStream_t stream)
{
    (void)in_sizes; (void)n_in; (void)out_size; (void)ws_size;

    const float* data = (const float*)d_in[0];
    const float* W1   = (const float*)d_in[1];   // (2,1,3,3,3)
    const float* W2   = (const float*)d_in[2];   // (2,2,3,3,3)
    const float* W3   = (const float*)d_in[3];
    const float* W4   = (const float*)d_in[4];
    const float* W5   = (const float*)d_in[5];   // (2,2,1,1,1)
    float* out        = (float*)d_out;           // [prob(NVOX) | regr(NVOX)]

    uint8_t* ws = (uint8_t*)d_ws;
    uint32_t* counter = (uint32_t*)ws;
    uint32_t* list    = (uint32_t*)(ws + 256);
    uint32_t* xA      = (uint32_t*)(ws + 256 + (size_t)4 * NVOX);
    uint32_t* xB      = xA + (NVOX + 1);

    k_reset<<<1, 32, 0, stream>>>(counter);
    k_scan<<<NVOX / 256, 256, 0, stream>>>(data, counter, list, xA, xB, out);

    const int CB = 2048, CT = 256;   // 16K waves grid-striding over tiles
    k_conv<0><<<CB, CT, 0, stream>>>(xA, (_Float16*)(xB + 1), nullptr, nullptr,
                                     counter, list, W1, nullptr, 1, 1);
    k_conv<0><<<CB, CT, 0, stream>>>(xB, (_Float16*)(xA + 1), nullptr, nullptr,
                                     counter, list, W2, nullptr, 2, 2);
    k_conv<0><<<CB, CT, 0, stream>>>(xA, (_Float16*)(xB + 1), nullptr, nullptr,
                                     counter, list, W3, nullptr, 2, 4);
    k_conv<1><<<CB, CT, 0, stream>>>(xB, nullptr, out, out + NVOX,
                                     counter, list, W4, W5, 2, 2);
}
